// BackboneGNN_17549236371683
// MI455X (gfx1250) — compile-verified, matrix operand order checked by reading, not run on hardware
//
#include <hip/hip_runtime.h>
#include <hip/hip_bf16.h>

// ---------------------------------------------------------------------------
// BackboneGNN for MI455X (gfx1250, wave32).
// Roofline: edge gather/scatter ~5 GB of HBM/L2 traffic dominates (~0.2 ms at
// 23.3 TB/s); the 14.7 GFLOP of GEMMs are WMMA-trivial. So:
//  - edge phase: wave-per-edge, float4/lane (128 feats = 32 lanes x 4), f32
//    atomics into an L2-resident 51 MB agg buffer (192 MB L2 keeps it hot).
//  - GEMMs: activations & weights pre-packed to f16 once (fused with the
//    scale/ReLU elementwise passes), then a pure load->v_wmma_f32_16x16x32_f16
//    loop. A tiles staged in LDS (pitch-padded, conflict-free ds_load_b128);
//    weight rows read straight from L1-resident f16 global.
// WMMA layouts per CDNA5 ISA 7.12.2:
//  A 16x32 f16: lane = row M; lanes<16 hold K {0..7,16..23}, lanes>=16 hold
//  K {8..15,24..31}. B mirrors A with lane = column N; since (A@W^T)[m,n]
//  needs B[k][n] = W[n][k], lane n reads W row n contiguously.
//  C/D: 8 f32/lane, column N = lane&15, rows M = i + 8*(lane>>4).
// ---------------------------------------------------------------------------

typedef __attribute__((ext_vector_type(16))) _Float16 v16h;
typedef __attribute__((ext_vector_type(8)))  _Float16 v8h;
typedef __attribute__((ext_vector_type(8)))  float    v8f;

#define NNODES 100000
#define NEDGES 1600000
#define DIM    128
#define NCLS   64
#define NGRAPH 128
#define BN_EPS 1e-5f
#define LDSPITCH 136   // halves per LDS row: 128 + 8 pad -> conflict-free b128

__global__ __launch_bounds__(256) void k_zero(float* __restrict__ p, int n) {
  int i = blockIdx.x * 256 + threadIdx.x;
  if (i < n) p[i] = 0.0f;
}

// histogram of an index array into float counters (in-degree / graph sizes)
__global__ __launch_bounds__(256) void k_count(const int* __restrict__ idx,
                                               float* __restrict__ cnt, int n) {
  int i = blockIdx.x * 256 + threadIdx.x;
  if (i < n) atomicAdd(&cnt[idx[i]], 1.0f);
}

// msg = relu(h[src] + xe); agg[dst] += msg.  One wave per edge, float4/lane.
__global__ __launch_bounds__(256) void k_edge_scatter(const float* __restrict__ h,
                                                      const float* __restrict__ xe,
                                                      const int* __restrict__ src,
                                                      const int* __restrict__ dst,
                                                      float* __restrict__ agg) {
  int e = blockIdx.x * 8 + (threadIdx.x >> 5);
  int lane = threadIdx.x & 31;
  if (e >= NEDGES) return;
  int s = src[e], d = dst[e];
  const float4 a = *(const float4*)(h  + (size_t)s * DIM + lane * 4);
  const float4 b = *(const float4*)(xe + (size_t)e * DIM + lane * 4);
  float* o = agg + (size_t)d * DIM + lane * 4;
  atomicAdd(o + 0, fmaxf(a.x + b.x, 0.0f));
  atomicAdd(o + 1, fmaxf(a.y + b.y, 0.0f));
  atomicAdd(o + 2, fmaxf(a.z + b.z, 0.0f));
  atomicAdd(o + 3, fmaxf(a.w + b.w, 0.0f));
}

// f32 -> f16 pack, optionally scaled by 1/max(cnt[row],1) and/or ReLU'd.
// (rows are DIM=128 wide when cnt is used)
__global__ __launch_bounds__(256) void k_pack_f16(const float* __restrict__ in,
                                                  _Float16* __restrict__ out,
                                                  const float* __restrict__ cnt,
                                                  int total, int relu_) {
  int i = blockIdx.x * 256 + threadIdx.x;
  if (i >= total) return;
  float v = in[i];
  if (cnt) v = v / fmaxf(cnt[i >> 7], 1.0f);
  if (relu_) v = fmaxf(v, 0.0f);
  out[i] = (_Float16)v;
}

// OUT[row0:row0+16, :OC] = A1@W1^T (+ A2@W2^T) + bias.  K = DIM = 128.
// block = (32, OC/16): one wave per 16-wide output column tile; A tiles
// staged in LDS once per block, weights streamed from f16 global (L1-hot).
__global__ __launch_bounds__(256) void k_gemm_wmma(const _Float16* __restrict__ A1,
                                                   const _Float16* __restrict__ W1,
                                                   const _Float16* __restrict__ A2,
                                                   const _Float16* __restrict__ W2,
                                                   const float* __restrict__ bias,
                                                   float* __restrict__ OUT,
                                                   int OC) {
  __shared__ _Float16 sA1[16 * LDSPITCH];
  __shared__ _Float16 sA2[16 * LDSPITCH];
  const int tid  = threadIdx.y * 32 + threadIdx.x;
  const int nthr = blockDim.y * 32;
  const int row0 = blockIdx.x * 16;

  // cooperative stage: 16 rows x 128 halves = 4KB per tile, 16B chunks
  for (int t = tid; t < 256; t += nthr) {
    int r = t >> 4, c = (t & 15) * 8;
    *(uint4*)(sA1 + r * LDSPITCH + c) =
        *(const uint4*)(A1 + (size_t)(row0 + r) * DIM + c);
    if (A2)
      *(uint4*)(sA2 + r * LDSPITCH + c) =
          *(const uint4*)(A2 + (size_t)(row0 + r) * DIM + c);
  }
  __syncthreads();

  const int lane = threadIdx.x;
  const int m    = lane & 15;
  const int hi   = lane >> 4;
  const int koff = hi * 8;
  const int col0 = threadIdx.y * 16;

  v8f acc = {};
  const _Float16* w1 = W1 + (size_t)(col0 + m) * DIM;
#pragma unroll
  for (int kc = 0; kc < DIM; kc += 32) {
    v8h a0 = *(const v8h*)(sA1 + m * LDSPITCH + kc + koff);
    v8h a1 = *(const v8h*)(sA1 + m * LDSPITCH + kc + 16 + koff);
    v8h b0 = *(const v8h*)(w1 + kc + koff);
    v8h b1 = *(const v8h*)(w1 + kc + 16 + koff);
    v16h va = __builtin_shufflevector(a0, a1, 0,1,2,3,4,5,6,7,8,9,10,11,12,13,14,15);
    v16h vb = __builtin_shufflevector(b0, b1, 0,1,2,3,4,5,6,7,8,9,10,11,12,13,14,15);
    acc = __builtin_amdgcn_wmma_f32_16x16x32_f16(false, va, false, vb,
                                                 (short)0, acc, false, false);
  }

  if (A2 != nullptr) {
    const _Float16* w2 = W2 + (size_t)(col0 + m) * DIM;
#pragma unroll
    for (int kc = 0; kc < DIM; kc += 32) {
      v8h a0 = *(const v8h*)(sA2 + m * LDSPITCH + kc + koff);
      v8h a1 = *(const v8h*)(sA2 + m * LDSPITCH + kc + 16 + koff);
      v8h b0 = *(const v8h*)(w2 + kc + koff);
      v8h b1 = *(const v8h*)(w2 + kc + 16 + koff);
      v16h va = __builtin_shufflevector(a0, a1, 0,1,2,3,4,5,6,7,8,9,10,11,12,13,14,15);
      v16h vb = __builtin_shufflevector(b0, b1, 0,1,2,3,4,5,6,7,8,9,10,11,12,13,14,15);
      acc = __builtin_amdgcn_wmma_f32_16x16x32_f16(false, va, false, vb,
                                                   (short)0, acc, false, false);
    }
  }

  const float b = bias ? bias[col0 + m] : 0.0f;
#pragma unroll
  for (int i = 0; i < 8; ++i) {
    int r = row0 + i + hi * 8;
    OUT[(size_t)r * OC + col0 + m] = acc[i] + b;
  }
}

// per-feature sum / sum-of-squares (block-partial, then f32 atomics)
__global__ __launch_bounds__(128) void k_bn_stats(const float* __restrict__ h,
                                                  float* __restrict__ sums) {
  int d = threadIdx.x;
  float s = 0.0f, ss = 0.0f;
  for (int n = blockIdx.x; n < NNODES; n += gridDim.x) {
    float v = h[(size_t)n * DIM + d];
    s += v; ss += v * v;
  }
  atomicAdd(&sums[d], s);
  atomicAdd(&sums[DIM + d], ss);
}

__global__ __launch_bounds__(256) void k_bn_apply(float* __restrict__ h,
                                                  const float* __restrict__ sums,
                                                  const float* __restrict__ gamma,
                                                  const float* __restrict__ beta) {
  int i = blockIdx.x * 256 + threadIdx.x;
  if (i >= NNODES * DIM) return;
  int d = i & (DIM - 1);
  float mean = sums[d] * (1.0f / NNODES);
  float var  = sums[DIM + d] * (1.0f / NNODES) - mean * mean;
  float v = gamma[d] * (h[i] - mean) * rsqrtf(var + BN_EPS) + beta[d];
  h[i] = fmaxf(v, 0.0f);
}

__global__ __launch_bounds__(256) void k_graph_sum(const float* __restrict__ h,
                                                   const int* __restrict__ batch,
                                                   float* __restrict__ g) {
  int i = blockIdx.x * 256 + threadIdx.x;
  if (i >= NNODES * DIM) return;
  int n = i >> 7, d = i & (DIM - 1);
  atomicAdd(&g[(size_t)batch[n] * DIM + d], h[i]);
}

__global__ __launch_bounds__(256) void k_graph_div(float* __restrict__ g,
                                                   const float* __restrict__ gcnt) {
  int i = blockIdx.x * 256 + threadIdx.x;
  if (i < NGRAPH * DIM) g[i] = g[i] / fmaxf(gcnt[i >> 7], 1.0f);
}

extern "C" void kernel_launch(void* const* d_in, const int* in_sizes, int n_in,
                              void* d_out, int out_size, void* d_ws, size_t ws_size,
                              hipStream_t stream) {
  (void)in_sizes; (void)n_in; (void)out_size; (void)ws_size;
  const float* x      = (const float*)d_in[0];
  const float* xe     = (const float*)d_in[1];
  const int*   eidx   = (const int*)d_in[2];
  const int*   batch  = (const int*)d_in[3];
  const float* Wl0    = (const float*)d_in[4];
  const float* bl0    = (const float*)d_in[5];
  const float* Wr0    = (const float*)d_in[6];
  const float* Wl1    = (const float*)d_in[7];
  const float* bl1    = (const float*)d_in[8];
  const float* Wr1    = (const float*)d_in[9];
  const float* gamma0 = (const float*)d_in[10];
  const float* beta0  = (const float*)d_in[11];
  const float* Wp     = (const float*)d_in[12];
  const float* bp     = (const float*)d_in[13];
  const int* src = eidx;
  const int* dst = eidx + NEDGES;

  const int ND = NNODES * DIM;                  // 12.8M
  float* ws    = (float*)d_ws;
  float* buf0  = ws;                            // [N,D] f32: agg, then h1
  float* h0    = buf0 + (size_t)ND;             // [N,D] f32
  float* cnt   = h0 + (size_t)ND;               // [N]
  float* bnsum = cnt + NNODES;                  // [2,D]
  float* gcnt  = bnsum + 2 * DIM;               // [G]
  _Float16* xh   = (_Float16*)(gcnt + NGRAPH);  // [N,D] f16
  _Float16* aggh = xh + (size_t)ND;             // [N,D] f16 (later relu(h1))
  _Float16* h0h  = aggh + (size_t)ND;           // [N,D] f16
  _Float16* wl0h = h0h + (size_t)ND;            // weight f16 packs
  _Float16* wr0h = wl0h + DIM * DIM;
  _Float16* wl1h = wr0h + DIM * DIM;
  _Float16* wr1h = wl1h + DIM * DIM;
  _Float16* wph  = wr1h + DIM * DIM;            // [C,D]

  float* out_h = (float*)d_out;                 // [N,C]
  float* out_g = out_h + (size_t)NNODES * NCLS; // [G,D]

  const int TB = 256;
  const int WDD = DIM * DIM, WCD = NCLS * DIM;
  dim3 gemm_blk(32, DIM / 16);                  // 8 waves
  dim3 proj_blk(32, NCLS / 16);                 // 4 waves

  // ---- one-time f16 packs of static operands ----
  k_pack_f16<<<(ND + TB - 1) / TB, TB, 0, stream>>>(x, xh, nullptr, ND, 0);
  k_pack_f16<<<(WDD + TB - 1) / TB, TB, 0, stream>>>(Wl0, wl0h, nullptr, WDD, 0);
  k_pack_f16<<<(WDD + TB - 1) / TB, TB, 0, stream>>>(Wr0, wr0h, nullptr, WDD, 0);
  k_pack_f16<<<(WDD + TB - 1) / TB, TB, 0, stream>>>(Wl1, wl1h, nullptr, WDD, 0);
  k_pack_f16<<<(WDD + TB - 1) / TB, TB, 0, stream>>>(Wr1, wr1h, nullptr, WDD, 0);
  k_pack_f16<<<(WCD + TB - 1) / TB, TB, 0, stream>>>(Wp, wph, nullptr, WCD, 0);

  // ---- in-degree counts (shared by both layers) ----
  k_zero<<<(NNODES + TB - 1) / TB, TB, 0, stream>>>(cnt, NNODES);
  k_count<<<(NEDGES + TB - 1) / TB, TB, 0, stream>>>(dst, cnt, NEDGES);

  // ---- layer 0: scatter-mean -> WMMA GEMM -> BN+ReLU ----
  k_zero<<<(ND + TB - 1) / TB, TB, 0, stream>>>(buf0, ND);
  k_edge_scatter<<<(NEDGES + 7) / 8, TB, 0, stream>>>(x, xe, src, dst, buf0);
  k_pack_f16<<<(ND + TB - 1) / TB, TB, 0, stream>>>(buf0, aggh, cnt, ND, 0);
  k_gemm_wmma<<<NNODES / 16, gemm_blk, 0, stream>>>(aggh, wl0h, xh, wr0h,
                                                    bl0, h0, DIM);
  k_zero<<<1, TB, 0, stream>>>(bnsum, 2 * DIM);
  k_bn_stats<<<512, DIM, 0, stream>>>(h0, bnsum);
  k_bn_apply<<<(ND + TB - 1) / TB, TB, 0, stream>>>(h0, bnsum, gamma0, beta0);
  k_pack_f16<<<(ND + TB - 1) / TB, TB, 0, stream>>>(h0, h0h, nullptr, ND, 0);

  // ---- layer 1: scatter-mean -> WMMA GEMM (h1 reuses buf0) ----
  k_zero<<<(ND + TB - 1) / TB, TB, 0, stream>>>(buf0, ND);
  k_edge_scatter<<<(NEDGES + 7) / 8, TB, 0, stream>>>(h0, xe, src, dst, buf0);
  k_pack_f16<<<(ND + TB - 1) / TB, TB, 0, stream>>>(buf0, aggh, cnt, ND, 0);
  k_gemm_wmma<<<NNODES / 16, gemm_blk, 0, stream>>>(aggh, wl1h, h0h, wr1h,
                                                    bl1, buf0, DIM);

  // ---- graph mean-pool over pre-projection features ----
  k_zero<<<(NGRAPH * DIM + TB - 1) / TB, TB, 0, stream>>>(out_g, NGRAPH * DIM);
  k_zero<<<1, TB, 0, stream>>>(gcnt, NGRAPH);
  k_count<<<(NNODES + TB - 1) / TB, TB, 0, stream>>>(batch, gcnt, NNODES);
  k_graph_sum<<<(ND + TB - 1) / TB, TB, 0, stream>>>(buf0, batch, out_g);
  k_graph_div<<<(NGRAPH * DIM + TB - 1) / TB, TB, 0, stream>>>(out_g, gcnt);

  // ---- node readout: relu(h1) @ Wp^T + bp  (pack reuses aggh) ----
  k_pack_f16<<<(ND + TB - 1) / TB, TB, 0, stream>>>(buf0, aggh, nullptr, ND, 1);
  k_gemm_wmma<<<NNODES / 16, proj_blk, 0, stream>>>(aggh, wph, nullptr, nullptr,
                                                    bp, out_h, NCLS);
}
